// LMUFFT_29635274342993
// MI455X (gfx1250) — matrix operand order repaired and checked by reading, not compiled, not run on hardware
//
#include <hip/hip_runtime.h>

#define B_   8
#define S_   4096
#define D_   128
#define M_   256
#define HID_ 256
#define F_   (M_ + D_)   // 384

typedef __bf16 bf16;
typedef __attribute__((ext_vector_type(16))) __bf16 v16bf;
typedef __attribute__((ext_vector_type(8)))  __bf16 v8bf;
typedef __attribute__((ext_vector_type(4)))  __bf16 v4bf;
typedef __attribute__((ext_vector_type(8)))  float  v8f;
typedef __attribute__((ext_vector_type(4)))  float  v4f;

// ---------------------------------------------------------------------------
// u = relu(x @ W_u + b_u); also write reversed, zero-padded bf16 copy ur so
// Toeplitz A-fragments in k_conv read ascending-contiguous and OOB lags hit
// the zero pad.  One wave32 per (b,s) row: 32 lanes x float4 = 128 = D.
// ---------------------------------------------------------------------------
__global__ void k_u(const float* __restrict__ x, const float* __restrict__ Wu,
                    const float* __restrict__ bu, bf16* __restrict__ ur) {
  const int lane = threadIdx.x & 31;
  const int row  = blockIdx.x * (blockDim.x >> 5) + (threadIdx.x >> 5);
  v4f xv = ((const v4f*)(x + (size_t)row * D_))[lane];
  v4f wv = ((const v4f*)Wu)[lane];
  float s = xv[0]*wv[0] + xv[1]*wv[1] + xv[2]*wv[2] + xv[3]*wv[3];
  // wave32 xor-butterfly reduction via ds_swizzle (group-of-32 xor masks)
  s += __int_as_float(__builtin_amdgcn_ds_swizzle(__float_as_int(s), 0x041f)); // xor 1
  s += __int_as_float(__builtin_amdgcn_ds_swizzle(__float_as_int(s), 0x081f)); // xor 2
  s += __int_as_float(__builtin_amdgcn_ds_swizzle(__float_as_int(s), 0x101f)); // xor 4
  s += __int_as_float(__builtin_amdgcn_ds_swizzle(__float_as_int(s), 0x201f)); // xor 8
  s += __int_as_float(__builtin_amdgcn_ds_swizzle(__float_as_int(s), 0x401f)); // xor 16
  if (lane == 0) {
    float uv = fmaxf(s + bu[0], 0.0f);
    int b  = row >> 12;          // row / S_
    int si = row & (S_ - 1);
    ur[b * (S_ + 64) + (S_ - 1 - si)] = (bf16)uv;
  }
}

// zero the 64-element tail pad of each batch's reversed-u row
__global__ void k_pad(bf16* __restrict__ ur) {
  int i = threadIdx.x + blockIdx.x * blockDim.x;
  if (i < B_ * 64) ur[(i >> 6) * (S_ + 64) + S_ + (i & 63)] = (bf16)0.0f;
}

// f32 -> bf16 bulk convert (float4 in, 4x bf16 out)
__global__ void k_cvt4(const float* __restrict__ src, bf16* __restrict__ dst,
                       int nvec) {
  int i = threadIdx.x + blockIdx.x * blockDim.x;
  if (i >= nvec) return;
  v4f v = ((const v4f*)src)[i];
  v4bf o;
  o[0] = (bf16)v[0]; o[1] = (bf16)v[1]; o[2] = (bf16)v[2]; o[3] = (bf16)v[3];
  ((v4bf*)dst)[i] = o;
}

// Wt[h, f] = bf16(W_h[f, h])  — transposed so B-fragments are contiguous in K
__global__ void k_wt(const float* __restrict__ Wh, bf16* __restrict__ Wt) {
  int i = threadIdx.x + blockIdx.x * blockDim.x;   // over HID_*F_
  if (i >= HID_ * F_) return;
  int h = i / F_, f = i - h * F_;
  Wt[i] = (bf16)Wh[f * HID_ + h];
}

// ---------------------------------------------------------------------------
// Causal Toeplitz conv as WMMA GEMM:
//   m[b, s0+r, q0+n] = sum_lag ur[b, S-1-(s0+r)+lag] * H[q0+n, lag]
// Workgroup = (b, s0) strip of 16 timesteps; 16 waves cover the 16 q-tiles.
// A fragment uses contiguous K order (as does B): element j  ->  K =
//   base+j, base = (lane>=16)*16.  With the reversed copy
// this is one ascending 16-element window, and lag>s reads land in the
// zero pad (so no masking, and the triangular K loop is exact).
// ---------------------------------------------------------------------------
__launch_bounds__(512)
__global__ void k_conv(const bf16* __restrict__ ur, const bf16* __restrict__ Hb,
                       bf16* __restrict__ mb) {
  const int s0   = blockIdx.x << 4;
  const int b    = blockIdx.y;
  const int q0   = (threadIdx.x >> 5) << 4;
  const int lane = threadIdx.x & 31;
  const int n    = lane & 15;      // A row r == B col n for this mapping
  const int half = lane >> 4;
  const int base = half << 4;      // A-matrix K base
  const int klo  = half << 4;      // B-matrix K base
  const bf16* __restrict__ Hrow = Hb + (size_t)(q0 + n) * S_ + klo;
  const bf16* __restrict__ urb  = ur + b * (S_ + 64);
  const int p0 = (S_ - 1) - s0 - n + base;
  v8f acc = {};
  const int nk = (s0 + 47) >> 5;   // lag blocks covering lag <= s0+15
  for (int kb = 0; kb < nk; ++kb) {
    const int lag0 = kb << 5;
    const bf16* up = urb + p0 + lag0;
    v16bf a;
#pragma unroll
    for (int j = 0; j < 8; ++j) { a[j] = up[j]; a[j + 8] = up[8 + j]; }
    v16bf bmat = *(const v16bf*)(Hrow + lag0);          // aligned 32B
    __builtin_prefetch(Hrow + lag0 + 64, 0, 1);         // H K-tile two steps ahead
    acc = __builtin_amdgcn_wmma_f32_16x16x32_bf16(
        false, a, false, bmat, (short)0, acc, false, false);
  }
  // C/D layout: VGPR i -> row = i + 8*half, col = n
  bf16* out = mb + ((size_t)(b * S_ + s0) * M_) + q0 + n;
#pragma unroll
  for (int i = 0; i < 8; ++i)
    out[(size_t)(i + (half << 3)) * M_] = (bf16)acc[i];
}

// ---------------------------------------------------------------------------
// h = relu([m, x] @ W_h + b_h), plus h_n = h[:, S-1, :].  K = 384 = 12 WMMA
// steps; K-blocks of 32 never straddle the m/x boundary (256 % 32 == 0).
// ---------------------------------------------------------------------------
__launch_bounds__(512)
__global__ void k_head(const bf16* __restrict__ mb, const bf16* __restrict__ xb,
                       const bf16* __restrict__ Wt, const float* __restrict__ bh,
                       float* __restrict__ h, float* __restrict__ hn) {
  const int s0   = blockIdx.x << 4;
  const int b    = blockIdx.y;
  const int hid0 = (threadIdx.x >> 5) << 4;
  const int lane = threadIdx.x & 31;
  const int n    = lane & 15;
  const int half = lane >> 4;
  const int base = half << 4;
  const int klo  = half << 4;
  const bf16* __restrict__ mrow = mb + (size_t)(b * S_ + s0 + n) * M_;
  const bf16* __restrict__ xrow = xb + (size_t)(b * S_ + s0 + n) * D_;
  const bf16* __restrict__ wrow = Wt + (size_t)(hid0 + n) * F_ + klo;
  v8f acc = {};
#pragma unroll
  for (int kb = 0; kb < F_ / 32; ++kb) {
    const int f0 = kb << 5;
    const int c1 = f0 + base;
    const int c2 = c1 + 8;
    v8bf lo = (c1 < M_) ? *(const v8bf*)(mrow + c1)
                        : *(const v8bf*)(xrow + (c1 - M_));
    v8bf hi = (c2 < M_) ? *(const v8bf*)(mrow + c2)
                        : *(const v8bf*)(xrow + (c2 - M_));
    v16bf a = __builtin_shufflevector(lo, hi, 0, 1, 2, 3, 4, 5, 6, 7,
                                      8, 9, 10, 11, 12, 13, 14, 15);
    v16bf bmat = *(const v16bf*)(wrow + f0);            // aligned 32B
    acc = __builtin_amdgcn_wmma_f32_16x16x32_bf16(
        false, a, false, bmat, (short)0, acc, false, false);
  }
  const float bias = bh[hid0 + n];
  float* out = h + (size_t)(b * S_ + s0) * HID_ + hid0 + n;
#pragma unroll
  for (int i = 0; i < 8; ++i) {
    const int row = i + (half << 3);
    float v = fmaxf(acc[i] + bias, 0.0f);
    out[(size_t)row * HID_] = v;
    if (s0 + row == S_ - 1) hn[b * HID_ + hid0 + n] = v;
  }
}

// ---------------------------------------------------------------------------
extern "C" void kernel_launch(void* const* d_in, const int* in_sizes, int n_in,
                              void* d_out, int out_size, void* d_ws,
                              size_t ws_size, hipStream_t stream) {
  (void)in_sizes; (void)n_in; (void)out_size; (void)ws_size;
  const float* x  = (const float*)d_in[0];
  const float* Wu = (const float*)d_in[1];
  const float* bu = (const float*)d_in[2];
  const float* Wh = (const float*)d_in[3];
  const float* bh = (const float*)d_in[4];
  const float* H  = (const float*)d_in[5];

  char* ws = (char*)d_ws;
  constexpr size_t o_ur = 0;                                   // B*(S+64)*2
  constexpr size_t o_x  = o_ur + ((size_t)B_ * (S_ + 64) * 2 + 255 & ~(size_t)255);
  constexpr size_t o_H  = o_x + (size_t)B_ * S_ * D_ * 2;      // x bf16
  constexpr size_t o_W  = o_H + (size_t)M_ * S_ * 2 + 256;     // +pad: tail K-block reads
  constexpr size_t o_m  = o_W + (size_t)HID_ * F_ * 2;
  bf16* ur_bf = (bf16*)(ws + o_ur);
  bf16* x_bf  = (bf16*)(ws + o_x);
  bf16* H_bf  = (bf16*)(ws + o_H);
  bf16* Wt_bf = (bf16*)(ws + o_W);
  bf16* m_bf  = (bf16*)(ws + o_m);                             // end ~27.6 MB

  float* h  = (float*)d_out;
  float* hn = h + (size_t)B_ * S_ * HID_;

  k_pad<<<1, 512, 0, stream>>>(ur_bf);
  k_u<<<(B_ * S_) / 8, 256, 0, stream>>>(x, Wu, bu, ur_bf);
  k_cvt4<<<(B_ * S_ * D_ / 4 + 255) / 256, 256, 0, stream>>>(x, x_bf,
                                                             B_ * S_ * D_ / 4);
  k_cvt4<<<(M_ * S_ / 4 + 255) / 256, 256, 0, stream>>>(H, H_bf, M_ * S_ / 4);
  k_wt<<<(HID_ * F_ + 255) / 256, 256, 0, stream>>>(Wh, Wt_bf);
  k_conv<<<dim3(S_ / 16, B_), 512, 0, stream>>>(ur_bf, H_bf, m_bf);
  k_head<<<dim3(S_ / 16, B_), 512, 0, stream>>>(m_bf, x_bf, Wt_bf, bh, h, hn);
}
